// AttentionGCN_39178691674118
// MI455X (gfx1250) — matrix operand order repaired and checked by reading, not compiled, not run on hardware
//
#include <hip/hip_runtime.h>
#include <hip/hip_bf16.h>
#include <math.h>

// ---------------------------------------------------------------------------
// STGCN-like pipeline on gfx1250 (wave32, WMMA 16x16x32 f16->f32).
// B=8, N=2048, F=64 -> 16384 independent (b,n) rows; row r = b*2048 + n.
//
// kernelW   : swizzle cw1 (f32) -> fragment-major f16 (one-time prep).
// kernelA   : fused tconv1 -> cheb(32x32) -> gated tconv2 (im2col K=96);
//             all matmuls via v_wmma_f32_16x16x32_f16. B operands are
//             pre-swizzled in LDS to per-lane fragment order (32B reads);
//             h2 written with vector b128 stores; per-row BN partials.
// kernelA2  : per-node BN stats (reduce 8 batches).
// kernelB   : node-BN+ReLU fused into A-packing of the 16x2048x16 matmul
//             against prepped cw1 (64 chained WMMA K-chunks).
// kernelC   : BN16 normalize + ReLU + 1x16 projection -> out (16384 f32).
// ---------------------------------------------------------------------------

typedef _Float16 v16h __attribute__((ext_vector_type(16)));
typedef _Float16 v8h  __attribute__((ext_vector_type(8)));
typedef float    v8f  __attribute__((ext_vector_type(8)));

#define NNODES 2048
#define NROWS  16384

// A-matrix 16x32 f16 fragment (ISA 7.12.2): lane = M row (mod 16);
// per lane the 16 elements are two contiguous 8-half runs:
// K = kb..kb+7 and kb+16..kb+23, kb = (lane<16) ? 0 : 8.
__device__ __forceinline__ v16h pack_a16(const _Float16* base, int stride) {
  const unsigned l = threadIdx.x & 31u;
  const int m  = l & 15;
  const int kb = (l < 16u) ? 0 : 8;
  const v8h* p = (const v8h*)(base + m * stride + kb);   // 16B aligned
  const v8h lo = p[0];   // K = kb .. kb+7
  const v8h hi = p[2];   // K = kb+16 .. kb+23
  return __builtin_shufflevector(lo, hi, 0, 1, 2, 3, 4, 5, 6, 7,
                                 8, 9, 10, 11, 12, 13, 14, 15);
}

__device__ __forceinline__ float sigmoidf_(float x) {
  return 1.0f / (1.0f + __expf(-x));
}

// ---------------------------------------------------------------------------
// cw1 (16x2048 f32, [o][kappa]) -> fragment-major f16: [ch][lane][e],
// lane = N col (mod 16); lane holds K = kb..kb+7, kb+16..kb+23, kb = 0 (l<16) / 8 (l>=16).
__global__ __launch_bounds__(256) void kernelW(const float* __restrict__ cw1,
                                               _Float16* __restrict__ cw1f) {
  const int idx = blockIdx.x * 256 + threadIdx.x;   // 0..32767
  const int ch  = idx >> 9;                          // K-chunk 0..63
  const int rem = idx & 511;
  const int l   = rem >> 4, e = rem & 15;
  const int o   = l & 15;
  const int kb  = (l < 16) ? 0 : 8;
  cw1f[idx] = (_Float16)cw1[o * 2048 + ch * 32 + kb + e + (e & 8)];
}

// ---------------------------------------------------------------------------
__global__ __launch_bounds__(128) void kernelA(
    const float* __restrict__ x,
    const float* __restrict__ w1, const float* __restrict__ b1,
    const float* __restrict__ w2, const float* __restrict__ b2,
    const float* __restrict__ w3, const float* __restrict__ b3,
    const float* __restrict__ chw, const float* __restrict__ chb,
    const float* __restrict__ t2w1, const float* __restrict__ t2b1,
    const float* __restrict__ t2w2, const float* __restrict__ t2b2,
    const float* __restrict__ t2w3, const float* __restrict__ t2b3,
    _Float16* __restrict__ h2buf, float* __restrict__ rowstat)
{
  __shared__ float xp_s[4][68];
  __shared__ __align__(32) _Float16 h1_s[4][80 * 32];  // 5 M-tiles (zero pad)
  __shared__ __align__(32) _Float16 g_s [4][68 * 32];  // 66 valid time rows
  __shared__ v16h w2f[3][3][2][32];   // [conv][kc][nt][lane] fragment-major
  __shared__ v16h chf[2][32];         // [nt][lane] fragment-major cheb W^T

  const int tid  = threadIdx.x;
  const int wid  = tid >> 5;
  const int lane = tid & 31;

  // ---- stage B operands pre-swizzled into per-lane fragment order ----
  {
    _Float16* wp = (_Float16*)w2f;    // 3*3*2*32*16 = 9216 halfs
    for (int idx = tid; idx < 9216; idx += 128) {
      const int conv = idx / 3072;
      const int rem  = idx % 3072;
      const int kc   = rem >> 10;             // /1024
      const int rem2 = rem & 1023;
      const int nt   = rem2 >> 9;             // /512
      const int rem3 = rem2 & 511;
      const int l    = rem3 >> 4, e = rem3 & 15;
      const int o    = nt * 16 + (l & 15);
      const int c    = ((l < 16) ? 0 : 8) + e + (e & 8);   // K within chunk = channel
      const float* src = (conv == 0) ? t2w1 : (conv == 1) ? t2w2 : t2w3;
      wp[idx] = (_Float16)src[o * 96 + c * 3 + kc];
    }
    _Float16* cp = (_Float16*)chf;    // 2*32*16 = 1024 halfs
    for (int idx = tid; idx < 1024; idx += 128) {
      const int nt   = idx >> 9;
      const int rem  = idx & 511;
      const int l    = rem >> 4, e = rem & 15;
      const int o    = nt * 16 + (l & 15);
      const int c    = ((l < 16) ? 0 : 8) + e + (e & 8);
      cp[idx] = (_Float16)chw[o * 32 + c];
    }
  }
  __syncthreads();

  const size_t r = (size_t)blockIdx.x * 4 + wid;   // 0..16383

  // ---- padded input row: xp[t], t in [0,68) ----
  float* xp = xp_s[wid];
  for (int t = lane; t < 68; t += 32)
    xp[t] = (t < 2 || t >= 66) ? 0.0f : x[r * 64 + (t - 2)];
  __syncthreads();

  // ---- tconv1: gated 1->32, T 68->66 (VALU; contraction is only 3) ----
  {
    const int c = lane;
    const float a0 = w1[c*3+0], a1 = w1[c*3+1], a2 = w1[c*3+2], ab = b1[c];
    const float c0 = w2[c*3+0], c1 = w2[c*3+1], c2 = w2[c*3+2], cb = b2[c];
    const float d0 = w3[c*3+0], d1 = w3[c*3+1], d2 = w3[c*3+2], db = b3[c];
    _Float16* h1 = h1_s[wid];
    for (int t = 0; t < 66; ++t) {
      const float u = a0*xp[t] + a1*xp[t+1] + a2*xp[t+2] + ab;
      const float v = c0*xp[t] + c1*xp[t+1] + c2*xp[t+2] + cb;
      const float w = d0*xp[t] + d1*xp[t+1] + d2*xp[t+2] + db;
      h1[t*32 + c] = (_Float16)fmaxf(u * sigmoidf_(v) + w, 0.0f);
    }
    for (int t = 66; t < 80; ++t) h1[t*32 + c] = (_Float16)0.0f;
  }
  __syncthreads();

  // ---- cheb: g = relu(h1 @ W^T + b); 66x32x32, B frags hoisted ----
  {
    _Float16* h1 = h1_s[wid];
    _Float16* g  = g_s[wid];
    const v16h B0 = chf[0][lane];
    const v16h B1 = chf[1][lane];
    const int o0 = lane & 15;
    const float cb0 = chb[o0], cb1v = chb[o0 + 16];
    const int off = (lane < 16) ? 0 : 8;
#pragma unroll
    for (int mt = 0; mt < 5; ++mt) {
      const v16h A = pack_a16(h1 + mt * 16 * 32, 32);
      v8f C0 = {}, C1 = {};
      C0 = __builtin_amdgcn_wmma_f32_16x16x32_f16(false, A, false, B0,
                                                  (short)0, C0, false, false);
      C1 = __builtin_amdgcn_wmma_f32_16x16x32_f16(false, A, false, B1,
                                                  (short)0, C1, false, false);
#pragma unroll
      for (int v = 0; v < 8; ++v) {
        const int t = mt * 16 + v + off;
        if (t < 66) {
          g[t * 32 + o0]      = (_Float16)fmaxf(C0[v] + cb0, 0.0f);
          g[t * 32 + o0 + 16] = (_Float16)fmaxf(C1[v] + cb1v, 0.0f);
        }
      }
    }
  }
  __syncthreads();

  // ---- tconv2: gated 32->32 k=3, T 66->64; im2col (A row shift = kc) ----
  float lsum = 0.0f, lsq = 0.0f;
  {
    _Float16* g = g_s[wid];
    const int off = (lane < 16) ? 0 : 8;
#pragma unroll
    for (int nt = 0; nt < 2; ++nt) {
      v16h Bq[3][3];
#pragma unroll
      for (int conv = 0; conv < 3; ++conv)
#pragma unroll
        for (int kc = 0; kc < 3; ++kc)
          Bq[conv][kc] = w2f[conv][kc][nt][lane];
      const int o = nt * 16 + (lane & 15);
      const float bb1 = t2b1[o], bb2 = t2b2[o], bb3 = t2b3[o];
#pragma unroll
      for (int mt = 0; mt < 4; ++mt) {
        v8f C1 = {}, C2 = {}, C3 = {};
#pragma unroll
        for (int kc = 0; kc < 3; ++kc) {
          const v16h A = pack_a16(g + (mt * 16 + kc) * 32, 32);  // g[t+kc][c]
          C1 = __builtin_amdgcn_wmma_f32_16x16x32_f16(false, A, false,
                   Bq[0][kc], (short)0, C1, false, false);
          C2 = __builtin_amdgcn_wmma_f32_16x16x32_f16(false, A, false,
                   Bq[1][kc], (short)0, C2, false, false);
          C3 = __builtin_amdgcn_wmma_f32_16x16x32_f16(false, A, false,
                   Bq[2][kc], (short)0, C3, false, false);
        }
        v8h hv;
#pragma unroll
        for (int v = 0; v < 8; ++v) {
          const float h = fmaxf((C1[v] + bb1) * sigmoidf_(C2[v] + bb2) +
                                (C3[v] + bb3), 0.0f);
          hv[v] = (_Float16)h;
          lsum += h; lsq += h * h;
        }
        // 8 consecutive t values -> single 16B store: [r][o*64 + t]
        *(v8h*)(h2buf + r * 2048 + (size_t)o * 64 + mt * 16 + off) = hv;
      }
    }
  }
  // per-row partials for node-BN (wave reduce, lane0 writes)
#pragma unroll
  for (int s = 16; s > 0; s >>= 1) {
    lsum += __shfl_down(lsum, s, 32);
    lsq  += __shfl_down(lsq,  s, 32);
  }
  if (lane == 0) { rowstat[r * 2] = lsum; rowstat[r * 2 + 1] = lsq; }
}

// ---------------------------------------------------------------------------
__global__ __launch_bounds__(256) void kernelA2(const float* __restrict__ rowstat,
                                                float* __restrict__ nstat) {
  const int n = blockIdx.x * 256 + threadIdx.x;   // 2048 nodes
  float s = 0.0f, q = 0.0f;
  for (int b = 0; b < 8; ++b) {
    const size_t rr = (size_t)b * NNODES + n;
    s += rowstat[rr * 2];
    q += rowstat[rr * 2 + 1];
  }
  const float inv = 1.0f / 16384.0f;              // 8*64*32 per node
  const float mean = s * inv;
  const float var  = q * inv - mean * mean;
  nstat[2 * n]     = mean;
  nstat[2 * n + 1] = rsqrtf(var + 1e-5f);
}

// ---------------------------------------------------------------------------
__global__ __launch_bounds__(256) void kernelB(
    const _Float16* __restrict__ h2buf, const _Float16* __restrict__ cw1f,
    const float* __restrict__ nstat,
    const float* __restrict__ bnn_g, const float* __restrict__ bnn_b,
    const float* __restrict__ cb1,
    float* __restrict__ ybuf, float* __restrict__ bn16sum,
    float* __restrict__ bn16sq)
{
  const int tid = threadIdx.x, wid = tid >> 5, lane = tid & 31;
  const int tile = blockIdx.x * 8 + wid;            // 1024 tiles of 16 rows
  const size_t rA = (size_t)tile * 16 + (lane & 15);
  const int n = (int)(rA & (NNODES - 1));
  const float mean = nstat[2 * n], rstd = nstat[2 * n + 1];
  const float sc = bnn_g[n] * rstd;
  const float sh = bnn_b[n] - mean * sc;            // relu(v*sc+sh) = node-BN
  const int kbA = (lane < 16) ? 0 : 8;
  const int o   = lane & 15;
  const v16h* bfrag = (const v16h*)cw1f;

  v8f acc = {};
  for (int ch = 0; ch < 64; ++ch) {                 // K = 2048 in 32-chunks
    const v8h* ap = (const v8h*)(h2buf + rA * 2048 + ch * 32 + kbA);
    const v8h a0 = ap[0];                           // K = kbA .. kbA+7
    const v8h a1 = ap[2];                           // K = kbA+16 .. kbA+23
    v16h A;
#pragma unroll
    for (int e = 0; e < 8; ++e) {
      A[e]     = (_Float16)fmaxf((float)a0[e] * sc + sh, 0.0f);
      A[e + 8] = (_Float16)fmaxf((float)a1[e] * sc + sh, 0.0f);
    }
    const v16h Bf = bfrag[ch * 32 + lane];          // pre-swizzled fragment
    acc = __builtin_amdgcn_wmma_f32_16x16x32_f16(false, A, false, Bf,
                                                 (short)0, acc, false, false);
  }
  const float cb = cb1[o];
  float ls = 0.0f, lq = 0.0f;
#pragma unroll
  for (int v = 0; v < 8; ++v) {
    const size_t rr = (size_t)tile * 16 + v + ((lane < 16) ? 0 : 8);
    const float y = acc[v] + cb;
    ybuf[rr * 16 + o] = y;
    ls += y; lq += y * y;
  }
  atomicAdd(&bn16sum[o], ls);                       // global_atomic_add_f32
  atomicAdd(&bn16sq[o],  lq);
}

// ---------------------------------------------------------------------------
__global__ __launch_bounds__(256) void kernelC(
    const float* __restrict__ ybuf, const float* __restrict__ bn16sum,
    const float* __restrict__ bn16sq, const float* __restrict__ g16,
    const float* __restrict__ b16, const float* __restrict__ cw2,
    const float* __restrict__ cb2, float* __restrict__ out)
{
  const size_t r = (size_t)blockIdx.x * 256 + threadIdx.x;   // 16384
  const float inv = 1.0f / (float)NROWS;
  float acc = cb2[0];
#pragma unroll
  for (int o = 0; o < 16; ++o) {
    const float mean = bn16sum[o] * inv;
    const float var  = bn16sq[o] * inv - mean * mean;
    const float rstd = rsqrtf(var + 1e-5f);
    const float t = (ybuf[r * 16 + o] - mean) * rstd * g16[o] + b16[o];
    acc += cw2[o] * fmaxf(t, 0.0f);
  }
  out[r] = acc;
}

// ---------------------------------------------------------------------------
extern "C" void kernel_launch(void* const* d_in, const int* in_sizes, int n_in,
                              void* d_out, int out_size, void* d_ws,
                              size_t ws_size, hipStream_t stream) {
  (void)in_sizes; (void)n_in; (void)out_size; (void)ws_size;
  const float* x      = (const float*)d_in[0];
  // d_in[1] edge_index, d_in[2] edge_weight: unused (ChebConv K=1)
  const float* tc1_w1 = (const float*)d_in[3];
  const float* tc1_b1 = (const float*)d_in[4];
  const float* tc1_w2 = (const float*)d_in[5];
  const float* tc1_b2 = (const float*)d_in[6];
  const float* tc1_w3 = (const float*)d_in[7];
  const float* tc1_b3 = (const float*)d_in[8];
  const float* cheb_w = (const float*)d_in[9];
  const float* cheb_b = (const float*)d_in[10];
  const float* tc2_w1 = (const float*)d_in[11];
  const float* tc2_b1 = (const float*)d_in[12];
  const float* tc2_w2 = (const float*)d_in[13];
  const float* tc2_b2 = (const float*)d_in[14];
  const float* tc2_w3 = (const float*)d_in[15];
  const float* tc2_b3 = (const float*)d_in[16];
  const float* bnn_g  = (const float*)d_in[17];
  const float* bnn_b  = (const float*)d_in[18];
  const float* cw1    = (const float*)d_in[19];
  const float* cb1    = (const float*)d_in[20];
  const float* bn16_g = (const float*)d_in[21];
  const float* bn16_b = (const float*)d_in[22];
  const float* cw2    = (const float*)d_in[23];
  const float* cb2    = (const float*)d_in[24];

  char* ws = (char*)d_ws;
  _Float16* h2buf = (_Float16*)ws; ws += (size_t)NROWS * 2048 * sizeof(_Float16);
  float* rowstat  = (float*)ws;    ws += (size_t)NROWS * 2 * sizeof(float);
  float* nstat    = (float*)ws;    ws += (size_t)NNODES * 2 * sizeof(float);
  float* ybuf     = (float*)ws;    ws += (size_t)NROWS * 16 * sizeof(float);
  float* bn16sum  = (float*)ws;    ws += 16 * sizeof(float);
  float* bn16sq   = (float*)ws;    ws += 16 * sizeof(float);
  _Float16* cw1f  = (_Float16*)ws; // 32768 halfs, fragment-major

  (void)hipMemsetAsync(bn16sum, 0, 32 * sizeof(float), stream);  // sum + sq

  kernelW<<<128, 256, 0, stream>>>(cw1, cw1f);
  kernelA<<<4096, 128, 0, stream>>>(x, tc1_w1, tc1_b1, tc1_w2, tc1_b2,
                                    tc1_w3, tc1_b3, cheb_w, cheb_b,
                                    tc2_w1, tc2_b1, tc2_w2, tc2_b2,
                                    tc2_w3, tc2_b3, h2buf, rowstat);
  kernelA2<<<8, 256, 0, stream>>>(rowstat, nstat);
  kernelB<<<128, 256, 0, stream>>>(h2buf, cw1f, nstat, bnn_g, bnn_b, cb1,
                                   ybuf, bn16sum, bn16sq);
  kernelC<<<64, 256, 0, stream>>>(ybuf, bn16sum, bn16sq, bn16_g, bn16_b,
                                  cw2, cb2, (float*)d_out);
}
